// OnnxMPNNLayer_16415365005578
// MI455X (gfx1250) — matrix operand order repaired and checked by reading, not compiled, not run on hardware
//
#include <hip/hip_runtime.h>
#include <hip/hip_bf16.h>

typedef __attribute__((ext_vector_type(16))) _Float16 v16h;
typedef __attribute__((ext_vector_type(8)))  _Float16 v8h;
typedef __attribute__((ext_vector_type(8)))  float    v8f;
typedef __attribute__((ext_vector_type(4)))  float    v4f;

#define HDIM 128

__device__ __forceinline__ v8f wmma_f16(v16h a, v16h b, v8f c) {
  return __builtin_amdgcn_wmma_f32_16x16x32_f16(
      /*neg_a=*/false, a, /*neg_b=*/false, b,
      /*c_mod=*/(short)0, c, /*reuse_a=*/false, /*reuse_b=*/false);
}

// Pack A-tile (16x32 f16) per-lane fragment from two contiguous 8-float runs.
// Lane layout used here (contiguous): halves 0..7 = K in [16h,16h+8), halves 8..15 = K in [16h+8,+8).
__device__ __forceinline__ v16h pack4(v4f r0, v4f r1, v4f r2, v4f r3) {
  v16h a;
  a[0]=(_Float16)r0.x;  a[1]=(_Float16)r0.y;  a[2]=(_Float16)r0.z;  a[3]=(_Float16)r0.w;
  a[4]=(_Float16)r1.x;  a[5]=(_Float16)r1.y;  a[6]=(_Float16)r1.z;  a[7]=(_Float16)r1.w;
  a[8]=(_Float16)r2.x;  a[9]=(_Float16)r2.y;  a[10]=(_Float16)r2.z; a[11]=(_Float16)r2.w;
  a[12]=(_Float16)r3.x; a[13]=(_Float16)r3.y; a[14]=(_Float16)r3.z; a[15]=(_Float16)r3.w;
  return a;
}

__device__ __forceinline__ v16h pack_a_f32(const float* run0, const float* run1) {
  return pack4(*(const v4f*)(run0), *(const v4f*)(run0 + 4),
               *(const v4f*)(run1), *(const v4f*)(run1 + 4));
}

// Non-temporal variant: for single-use streaming data (edge_attr) -> TH=NT,
// protects x's residency in the 192MB L2 that the src/dst gathers rely on.
__device__ __forceinline__ v16h pack_a_f32_nt(const float* run0, const float* run1) {
  return pack4(__builtin_nontemporal_load((const v4f*)(run0)),
               __builtin_nontemporal_load((const v4f*)(run0 + 4)),
               __builtin_nontemporal_load((const v4f*)(run1)),
               __builtin_nontemporal_load((const v4f*)(run1 + 4)));
}

__device__ __forceinline__ v16h pack_a_f16(const _Float16* p0, const _Float16* p1) {
  v8h lo = *(const v8h*)p0;
  v8h hi = *(const v8h*)p1;
  v16h a;
#pragma unroll
  for (int i = 0; i < 8; ++i) { a[i] = lo[i]; a[8 + i] = hi[i]; }
  return a;
}

__device__ __forceinline__ float sigmoidf_(float t) {
  return 1.0f / (1.0f + __expf(-t));
}

#define V8F_ZERO ((v8f){0,0,0,0,0,0,0,0})

// ---------------------------------------------------------------------------
// Kernel 0: zero the aggregation buffer
// ---------------------------------------------------------------------------
__global__ void zero_kernel(float* __restrict__ p, long long n) {
  long long i = (long long)blockIdx.x * blockDim.x + threadIdx.x;
  long long stride = (long long)gridDim.x * blockDim.x;
  for (; i < n; i += stride) p[i] = 0.0f;
}

// ---------------------------------------------------------------------------
// Kernel 1: per-edge MLP + atomic scatter by dst.
// One wave32 per 32-edge tile (two 16-row sub-tiles): each B tile fetched from
// LDS feeds TWO WMMAs back-to-back, with the next B tile prefetched before the
// current pair issues, hiding ds latency inside the matrix pipe.
// ---------------------------------------------------------------------------
__global__ void __launch_bounds__(128)
edge_mlp_kernel(const float* __restrict__ x,
                const long long* __restrict__ ei,   // [2,E]: row0=src, row1=dst
                const float* __restrict__ ea,       // [E,128]
                const float* __restrict__ W1,       // [128,384] row-major
                const float* __restrict__ b1,       // [128]
                const float* __restrict__ W2,       // [128,128] row-major
                const float* __restrict__ b2,       // [128]
                float* __restrict__ agg,            // [N,128] (atomic dst)
                int E, int ntiles) {
  extern __shared__ __align__(32) char smem[];
  _Float16* sW1 = (_Float16*)smem;            // 128*384 f16 = 96 KB
  _Float16* sW2 = sW1 + 128 * 384;            // 128*128 f16 = 32 KB
  _Float16* sH1 = sW2 + 128 * 128;            // 4 waves * 2*16*128 f16 = 32 KB

  for (int i = threadIdx.x; i < 128 * 384; i += blockDim.x) sW1[i] = (_Float16)W1[i];
  for (int i = threadIdx.x; i < 128 * 128; i += blockDim.x) sW2[i] = (_Float16)W2[i];
  __syncthreads();

  const int wave = threadIdx.x >> 5;
  const int lane = threadIdx.x & 31;
  const int tile = blockIdx.x * 4 + wave;
  if (tile >= ntiles) return;                 // wave-uniform: EXEC stays full

  const int h = lane >> 4;
  const int m = lane & 15;
  int e0 = tile * 32 + m;       if (e0 >= E) e0 = E - 1;
  int e1 = tile * 32 + 16 + m;  if (e1 >= E) e1 = E - 1;
  const float* xs0  = x + ei[e0] * HDIM;
  const float* xd0  = x + ei[(long long)E + e0] * HDIM;
  const float* ea0  = ea + (long long)e0 * HDIM;
  const float* xs1  = x + ei[e1] * HDIM;
  const float* xd1  = x + ei[(long long)E + e1] * HDIM;
  const float* ea1  = ea + (long long)e1 * HDIM;

  // ---- layer 1: [32x384] @ [384x128] ----
  v8f acc0[8], acc1[8];
#pragma unroll
  for (int nt = 0; nt < 8; ++nt) { acc0[nt] = V8F_ZERO; acc1[nt] = V8F_ZERO; }

#pragma unroll
  for (int kt = 0; kt < 12; ++kt) {
    v16h a0, a1;
    if (kt < 4) {
      int kb = kt * 32;
      a0 = pack_a_f32(xs0 + kb + h * 16, xs0 + kb + 8 + h * 16);
      a1 = pack_a_f32(xs1 + kb + h * 16, xs1 + kb + 8 + h * 16);
    } else if (kt < 8) {
      int kb = (kt - 4) * 32;
      a0 = pack_a_f32(xd0 + kb + h * 16, xd0 + kb + 8 + h * 16);
      a1 = pack_a_f32(xd1 + kb + h * 16, xd1 + kb + 8 + h * 16);
    } else {
      int kb = (kt - 8) * 32;
      a0 = pack_a_f32_nt(ea0 + kb + h * 16, ea0 + kb + 8 + h * 16);  // streamed once
      a1 = pack_a_f32_nt(ea1 + kb + h * 16, ea1 + kb + 8 + h * 16);
    }
    v16h b = *(const v16h*)(sW1 + m * 384 + kt * 32 + h * 16);   // nt=0
#pragma unroll
    for (int nt = 0; nt < 8; ++nt) {
      v16h bn;
      if (nt < 7) bn = *(const v16h*)(sW1 + ((nt + 1) * 16 + m) * 384 + kt * 32 + h * 16);
      acc0[nt] = wmma_f16(a0, b, acc0[nt]);
      acc1[nt] = wmma_f16(a1, b, acc1[nt]);
      if (nt < 7) b = bn;
    }
  }

  // ---- bias + ReLU, stage h1 (2 x 16x128 f16) into per-wave LDS slab ----
  _Float16* myH1 = sH1 + wave * 2 * 16 * 128;
#pragma unroll
  for (int nt = 0; nt < 8; ++nt) {
    float bb = b1[nt * 16 + m];
#pragma unroll
    for (int v = 0; v < 8; ++v) {
      float v0 = acc0[nt][v] + bb; v0 = v0 > 0.0f ? v0 : 0.0f;
      float v1 = acc1[nt][v] + bb; v1 = v1 > 0.0f ? v1 : 0.0f;
      myH1[(h * 8 + v) * 128 + nt * 16 + m] = (_Float16)v0;
      myH1[2048 + (h * 8 + v) * 128 + nt * 16 + m] = (_Float16)v1;
    }
  }
  asm volatile("s_wait_dscnt 0x0" ::: "memory");   // same-wave LDS store->load

  // ---- layer 2: [32x128] @ [128x128] ----
#pragma unroll
  for (int nt = 0; nt < 8; ++nt) { acc0[nt] = V8F_ZERO; acc1[nt] = V8F_ZERO; }
#pragma unroll
  for (int kt = 0; kt < 4; ++kt) {
    const _Float16* base0 = myH1 + m * 128 + kt * 32;
    const _Float16* base1 = myH1 + 2048 + m * 128 + kt * 32;
    v16h a0 = pack_a_f16(base0 + h * 16, base0 + 8 + h * 16);
    v16h a1 = pack_a_f16(base1 + h * 16, base1 + 8 + h * 16);
    v16h b = *(const v16h*)(sW2 + m * 128 + kt * 32 + h * 16);
#pragma unroll
    for (int nt = 0; nt < 8; ++nt) {
      v16h bn;
      if (nt < 7) bn = *(const v16h*)(sW2 + ((nt + 1) * 16 + m) * 128 + kt * 32 + h * 16);
      acc0[nt] = wmma_f16(a0, b, acc0[nt]);
      acc1[nt] = wmma_f16(a1, b, acc1[nt]);
      if (nt < 7) b = bn;
    }
  }

  // ---- bias + atomic scatter into agg[dst] ----
#pragma unroll
  for (int nt = 0; nt < 8; ++nt) {
    float bb = b2[nt * 16 + m];
#pragma unroll
    for (int v = 0; v < 8; ++v) {
      int er0 = tile * 32 + h * 8 + v;
      int er1 = tile * 32 + 16 + h * 8 + v;
      if (er0 < E) {
        long long dr = ei[(long long)E + er0];
        unsafeAtomicAdd(&agg[dr * HDIM + nt * 16 + m], acc0[nt][v] + bb);
      }
      if (er1 < E) {
        long long dr = ei[(long long)E + er1];
        unsafeAtomicAdd(&agg[dr * HDIM + nt * 16 + m], acc1[nt][v] + bb);
      }
    }
  }
}

// ---------------------------------------------------------------------------
// Kernel 2: GRU cell. One wave per 32-node tile; each LDS B tile feeds 2 WMMAs.
// ---------------------------------------------------------------------------
__device__ __forceinline__ void gemm128x2(const v16h* A0, const v16h* A1,
                                          const _Float16* sW, int n, int h,
                                          v8f& c0, v8f& c1) {
  v16h b = *(const v16h*)(sW + n * 128 + 0 * 32 + h * 16);
#pragma unroll
  for (int kt = 0; kt < 4; ++kt) {
    v16h bn;
    if (kt < 3) bn = *(const v16h*)(sW + n * 128 + (kt + 1) * 32 + h * 16);
    c0 = wmma_f16(A0[kt], b, c0);
    c1 = wmma_f16(A1[kt], b, c1);
    if (kt < 3) b = bn;
  }
}

__global__ void __launch_bounds__(256)
gru_kernel(const float* __restrict__ x,
           const float* __restrict__ agg,
           const float* __restrict__ Wih,  // [384,128] row-major
           const float* __restrict__ bih,  // [384]
           const float* __restrict__ Whh,  // [384,128]
           const float* __restrict__ bhh,  // [384]
           float* __restrict__ out,        // [N,128]
           int N, int ntiles) {
  extern __shared__ __align__(32) char smem[];
  _Float16* sWih = (_Float16*)smem;          // 384*128 f16 = 96 KB
  _Float16* sWhh = sWih + 384 * 128;         // 96 KB

  for (int i = threadIdx.x; i < 384 * 128; i += blockDim.x) {
    sWih[i] = (_Float16)Wih[i];
    sWhh[i] = (_Float16)Whh[i];
  }
  __syncthreads();

  const int wave = threadIdx.x >> 5;
  const int lane = threadIdx.x & 31;
  const int tile = blockIdx.x * 8 + wave;
  if (tile >= ntiles) return;

  const int h = lane >> 4;
  const int m = lane & 15;
  int n0 = tile * 32 + m;      if (n0 >= N) n0 = N - 1;
  int n1 = tile * 32 + 16 + m; if (n1 >= N) n1 = N - 1;

  // A fragments: aggregated (GRU input) and x (GRU hidden) for both sub-tiles
  v16h Aa0[4], Ax0[4], Aa1[4], Ax1[4];
#pragma unroll
  for (int kt = 0; kt < 4; ++kt) {
    const float* a0p = agg + (long long)n0 * HDIM + kt * 32;
    const float* x0p = x   + (long long)n0 * HDIM + kt * 32;
    const float* a1p = agg + (long long)n1 * HDIM + kt * 32;
    const float* x1p = x   + (long long)n1 * HDIM + kt * 32;
    Aa0[kt] = pack_a_f32(a0p + h * 16, a0p + 8 + h * 16);
    Ax0[kt] = pack_a_f32(x0p + h * 16, x0p + 8 + h * 16);
    Aa1[kt] = pack_a_f32(a1p + h * 16, a1p + 8 + h * 16);
    Ax1[kt] = pack_a_f32(x1p + h * 16, x1p + 8 + h * 16);
  }

#pragma unroll
  for (int j = 0; j < 8; ++j) {
    const int nc = j * 16 + m;               // output column within H

    // r gate
    v8f gi0 = V8F_ZERO, gi1 = V8F_ZERO, gh0 = V8F_ZERO, gh1 = V8F_ZERO;
    gemm128x2(Aa0, Aa1, sWih, nc, h, gi0, gi1);
    gemm128x2(Ax0, Ax1, sWhh, nc, h, gh0, gh1);
    float bi = bih[nc], bh = bhh[nc];
    v8f r0, r1;
#pragma unroll
    for (int v = 0; v < 8; ++v) {
      r0[v] = sigmoidf_(gi0[v] + bi + gh0[v] + bh);
      r1[v] = sigmoidf_(gi1[v] + bi + gh1[v] + bh);
    }

    // z gate
    gi0 = V8F_ZERO; gi1 = V8F_ZERO; gh0 = V8F_ZERO; gh1 = V8F_ZERO;
    gemm128x2(Aa0, Aa1, sWih, 128 + nc, h, gi0, gi1);
    gemm128x2(Ax0, Ax1, sWhh, 128 + nc, h, gh0, gh1);
    bi = bih[128 + nc]; bh = bhh[128 + nc];
    v8f z0, z1;
#pragma unroll
    for (int v = 0; v < 8; ++v) {
      z0[v] = sigmoidf_(gi0[v] + bi + gh0[v] + bh);
      z1[v] = sigmoidf_(gi1[v] + bi + gh1[v] + bh);
    }

    // n gate + output (non-temporal stores: out is write-once)
    gi0 = V8F_ZERO; gi1 = V8F_ZERO; gh0 = V8F_ZERO; gh1 = V8F_ZERO;
    gemm128x2(Aa0, Aa1, sWih, 256 + nc, h, gi0, gi1);
    gemm128x2(Ax0, Ax1, sWhh, 256 + nc, h, gh0, gh1);
    bi = bih[256 + nc]; bh = bhh[256 + nc];
#pragma unroll
    for (int v = 0; v < 8; ++v) {
      int r0w = tile * 32 + h * 8 + v;
      int r1w = tile * 32 + 16 + h * 8 + v;
      if (r0w < N) {
        float nn = tanhf(gi0[v] + bi + r0[v] * (gh0[v] + bh));
        float xv = x[(long long)r0w * HDIM + nc];
        __builtin_nontemporal_store((1.0f - z0[v]) * nn + z0[v] * xv,
                                    &out[(long long)r0w * HDIM + nc]);
      }
      if (r1w < N) {
        float nn = tanhf(gi1[v] + bi + r1[v] * (gh1[v] + bh));
        float xv = x[(long long)r1w * HDIM + nc];
        __builtin_nontemporal_store((1.0f - z1[v]) * nn + z1[v] * xv,
                                    &out[(long long)r1w * HDIM + nc]);
      }
    }
  }
}

// ---------------------------------------------------------------------------
extern "C" void kernel_launch(void* const* d_in, const int* in_sizes, int n_in,
                              void* d_out, int out_size, void* d_ws, size_t ws_size,
                              hipStream_t stream) {
  const float*     x   = (const float*)d_in[0];
  const long long* ei  = (const long long*)d_in[1];
  const float*     ea  = (const float*)d_in[2];
  const float*     W1  = (const float*)d_in[3];
  const float*     b1  = (const float*)d_in[4];
  const float*     W2  = (const float*)d_in[5];
  const float*     b2  = (const float*)d_in[6];
  const float*     Wih = (const float*)d_in[7];
  const float*     bih = (const float*)d_in[8];
  const float*     Whh = (const float*)d_in[9];
  const float*     bhh = (const float*)d_in[10];
  float* out = (float*)d_out;
  float* agg = (float*)d_ws;

  const int N = in_sizes[0] / HDIM;
  const int E = in_sizes[2] / HDIM;

  // 0) zero aggregation buffer
  {
    long long n = (long long)N * HDIM;
    zero_kernel<<<2048, 256, 0, stream>>>(agg, n);
  }

  // 1) edge MLP + scatter-add (32 edges per wave, 4 waves per block)
  {
    int ntiles = (E + 31) / 32;
    int grid   = (ntiles + 3) / 4;
    size_t lds = (size_t)(128 * 384 + 128 * 128 + 4 * 2 * 16 * 128) * 2;  // 160 KB
    edge_mlp_kernel<<<grid, 128, lds, stream>>>(x, ei, ea, W1, b1, W2, b2,
                                                agg, E, ntiles);
  }

  // 2) GRU cell (32 nodes per wave, 8 waves per block)
  {
    int ntiles = (N + 31) / 32;
    int grid   = (ntiles + 7) / 8;
    size_t lds = (size_t)(2 * 384 * 128) * 2;            // 192 KB
    gru_kernel<<<grid, 256, lds, stream>>>(x, agg, Wih, bih, Whh, bhh,
                                           out, N, ntiles);
  }
}
